// MB_HGTBot_32031866093623
// MI455X (gfx1250) — matrix operand order repaired and checked by reading, not compiled, not run on hardware
//
#include <hip/hip_runtime.h>
#include <hip/hip_bf16.h>

typedef __attribute__((ext_vector_type(16))) _Float16 v16h;
typedef __attribute__((ext_vector_type(8)))  float    v8f;

#define NEG_SLOPE 0.01f
#define LN_EPS 1e-5f

// ---------------- pack W [K,Nout] (or transposed X[Nout,K]) into WMMA B-fragment layout ----------------
// packed index: (((tile_n*KC + kc)*32 + lane)*16 + j) ; col = tile_n*16 + (lane&15), kk = kc*32 + (lane>>4)*16 + j
__global__ void pack_b_kernel(const float* __restrict__ W, _Float16* __restrict__ out,
                              int K, int Nout, int transposed)
{
    int idx = blockIdx.x * blockDim.x + threadIdx.x;
    int total = K * Nout;
    if (idx >= total) return;
    int j    = idx & 15;
    int t    = idx >> 4;
    int lane = t & 31;
    t >>= 5;
    int KC   = K >> 5;
    int kc   = t % KC;
    int tn   = t / KC;
    int col  = tn * 16 + (lane & 15);
    int kk   = kc * 32 + (lane >> 4) * 16 + j;
    float v  = transposed ? W[(size_t)col * K + kk] : W[(size_t)kk * Nout + col];
    out[idx] = (_Float16)v;
}

// ---------------- A-fragment load: f32 row-major -> 16x32 f16 fragment (contiguous k = 16*half + j) ----------------
__device__ __forceinline__ v16h load_a_frag(const float* arow)
{
    const float4* p0 = reinterpret_cast<const float4*>(arow);
    float4 f0 = p0[0], f1 = p0[1];
    const float4* p1 = reinterpret_cast<const float4*>(arow + 8);
    float4 f2 = p1[0], f3 = p1[1];
    v16h a;
    a[0]=(_Float16)f0.x;  a[1]=(_Float16)f0.y;  a[2]=(_Float16)f0.z;  a[3]=(_Float16)f0.w;
    a[4]=(_Float16)f1.x;  a[5]=(_Float16)f1.y;  a[6]=(_Float16)f1.z;  a[7]=(_Float16)f1.w;
    a[8]=(_Float16)f2.x;  a[9]=(_Float16)f2.y;  a[10]=(_Float16)f2.z; a[11]=(_Float16)f2.w;
    a[12]=(_Float16)f3.x; a[13]=(_Float16)f3.y; a[14]=(_Float16)f3.z; a[15]=(_Float16)f3.w;
    return a;
}

// ---------------- generic WMMA GEMM: C[M,*] = act(A[M,K] @ Wpacked + bias) ----------------
// one wave computes a 16 x (16*TN) strip (A-fragment reused TN times); 4 waves/block along M
template<int TN, bool LEAKY>
__global__ void gemm_wmma_kernel(const float* __restrict__ A,
                                 const _Float16* __restrict__ Bp,
                                 const float* __restrict__ bias,
                                 float* __restrict__ C, int ldc, int coloff,
                                 int mtiles, int K)
{
    const int wave = threadIdx.x >> 5;
    const int lane = threadIdx.x & 31;
    const int tm   = blockIdx.x * 4 + wave;
    const int tn0  = blockIdx.y * TN;
    if (tm >= mtiles) return;                 // wave-uniform: EXEC stays all-ones for WMMA
    const int rl   = lane & 15;
    const int half = lane >> 4;
    const int KC   = K >> 5;
    const float*    arow  = A + (size_t)(tm * 16 + rl) * K + half * 16;
    const _Float16* bbase = Bp + (size_t)lane * 16;
    v8f acc[TN];
    #pragma unroll
    for (int t = 0; t < TN; ++t) acc[t] = {};
    for (int kc = 0; kc < KC; ++kc) {
        v16h a = load_a_frag(arow + kc * 32);
        #pragma unroll
        for (int t = 0; t < TN; ++t) {
            v16h b = *reinterpret_cast<const v16h*>(bbase + ((size_t)(tn0 + t) * KC + kc) * 512);
            acc[t] = __builtin_amdgcn_wmma_f32_16x16x32_f16(false, a, false, b, (short)0, acc[t], false, false);
        }
    }
    const int rbase = tm * 16 + half * 8;
    #pragma unroll
    for (int t = 0; t < TN; ++t) {
        const int col = (tn0 + t) * 16 + rl;
        float bb = bias ? bias[col] : 0.0f;
        float* cp = C + (size_t)rbase * ldc + coloff + col;
        #pragma unroll
        for (int v = 0; v < 8; ++v) {
            float val = acc[t][v] + bb;
            if (LEAKY) val = val >= 0.0f ? val : NEG_SLOPE * val;
            cp[(size_t)v * ldc] = val;
        }
    }
}

// ---------------- tiny-K embed branches (K=5, K=3): scalar ----------------
__global__ void embed_small_kernel(const float* __restrict__ in, int K,
                                   const float* __restrict__ W, const float* __restrict__ b,
                                   float* __restrict__ x, int coloff, int M)
{
    int idx = blockIdx.x * blockDim.x + threadIdx.x;
    if (idx >= M * 48) return;
    int row = idx / 48, c = idx % 48;
    float s = b[c];
    for (int k = 0; k < K; ++k) s += in[(size_t)row * K + k] * W[k * 48 + c];
    s = s >= 0.0f ? s : NEG_SLOPE * s;
    x[(size_t)row * 192 + coloff + c] = s;
}

// ---------------- fused residual + LayerNorm (+ optional BN affine) ----------------
template<bool BN>
__global__ void add_ln_kernel(const float* __restrict__ X, const float* __restrict__ R,
                              const float* __restrict__ g, const float* __restrict__ b,
                              const float* __restrict__ bng, const float* __restrict__ bnb,
                              float* __restrict__ out)
{
    __shared__ float sbuf[192];
    __shared__ float sm, sv;
    const int row = blockIdx.x, t = threadIdx.x;
    const size_t base = (size_t)row * 192 + t;
    float y = X[base] + R[base];
    sbuf[t] = y;
    __syncthreads();
    if (t < 64) sbuf[t] += sbuf[t + 128];
    __syncthreads();
    for (int s = 64; s > 0; s >>= 1) { if (t < s) sbuf[t] += sbuf[t + s]; __syncthreads(); }
    if (t == 0) sm = sbuf[0] * (1.0f / 192.0f);
    __syncthreads();
    float m = sm;
    float d = y - m;
    sbuf[t] = d * d;
    __syncthreads();
    if (t < 64) sbuf[t] += sbuf[t + 128];
    __syncthreads();
    for (int s = 64; s > 0; s >>= 1) { if (t < s) sbuf[t] += sbuf[t + s]; __syncthreads(); }
    if (t == 0) sv = sbuf[0] * (1.0f / 192.0f);
    __syncthreads();
    float val = d * rsqrtf(sv + LN_EPS) * g[t] + b[t];
    if (BN) val = val * (bng[t] * rsqrtf(1.0f + LN_EPS)) + bnb[t];
    out[base] = val;
}

// ---------------- row L2 normalize ----------------
__global__ void normalize_kernel(const float* __restrict__ Hh, float* __restrict__ out)
{
    __shared__ float sbuf[192];
    __shared__ float sn;
    const int row = blockIdx.x, t = threadIdx.x;
    const size_t base = (size_t)row * 192 + t;
    float y = Hh[base];
    sbuf[t] = y * y;
    __syncthreads();
    if (t < 64) sbuf[t] += sbuf[t + 128];
    __syncthreads();
    for (int s = 64; s > 0; s >>= 1) { if (t < s) sbuf[t] += sbuf[t + s]; __syncthreads(); }
    if (t == 0) sn = fmaxf(sqrtf(sbuf[0]), 1e-12f);
    __syncthreads();
    out[base] = y / sn;
}

// ---------------- RGCN edge ops ----------------
__global__ void zero_kernel(float* __restrict__ p, size_t n)
{
    size_t i = (size_t)blockIdx.x * blockDim.x + threadIdx.x;
    if (i < n) p[i] = 0.0f;
}

__global__ void edge_count_kernel(const int* __restrict__ dst, const int* __restrict__ et,
                                  float* __restrict__ cnt, int E, int Nn)
{
    int e = blockIdx.x * blockDim.x + threadIdx.x;
    if (e >= E) return;
    atomicAdd(&cnt[(size_t)et[e] * Nn + dst[e]], 1.0f);
}

__global__ void edge_scatter_kernel(const int* __restrict__ src, const int* __restrict__ dst,
                                    const int* __restrict__ et,
                                    const float* __restrict__ m0, const float* __restrict__ m1,
                                    float* __restrict__ acc, int E, int Nn)
{
    size_t idx = (size_t)blockIdx.x * blockDim.x + threadIdx.x;
    if (idx >= (size_t)E * 48) return;
    int e = (int)(idx / 48);
    int c = (int)(idx % 48) * 4;
    int r = et[e];
    const float* m = r ? m1 : m0;
    float4 v = *reinterpret_cast<const float4*>(m + (size_t)src[e] * 192 + c);
    float* pp = acc + (size_t)r * Nn * 192 + (size_t)dst[e] * 192 + c;
    atomicAdd(pp + 0, v.x); atomicAdd(pp + 1, v.y);
    atomicAdd(pp + 2, v.z); atomicAdd(pp + 3, v.w);
}

__global__ void rgcn_combine_kernel(const float* __restrict__ root,
                                    const float* __restrict__ acc, const float* __restrict__ cnt,
                                    float* __restrict__ out, int Nn)
{
    size_t idx = (size_t)blockIdx.x * blockDim.x + threadIdx.x;
    if (idx >= (size_t)Nn * 192) return;
    int row = (int)(idx / 192);
    float c0 = fmaxf(cnt[row], 1.0f);
    float c1 = fmaxf(cnt[Nn + row], 1.0f);
    out[idx] = root[idx] + acc[idx] / c0 + acc[(size_t)Nn * 192 + idx] / c1;
}

// ---------------- final 192->2 head ----------------
__global__ void head_out_kernel(const float* __restrict__ h2, const float* __restrict__ W3,
                                const float* __restrict__ b3, float* __restrict__ out, int M)
{
    int row = blockIdx.x * blockDim.x + threadIdx.x;
    if (row >= M) return;
    float s0 = b3[0], s1 = b3[1];
    for (int k = 0; k < 192; ++k) {
        float v = h2[(size_t)row * 192 + k];
        s0 += v * W3[2 * k];
        s1 += v * W3[2 * k + 1];
    }
    out[2 * (size_t)row]     = s0;
    out[2 * (size_t)row + 1] = s1;
}

// ---------------- contrastive loss: WMMA sim tiles + prefetch, fused exp/mask/reduce ----------------
// grid.x = 625 row tiles, grid.y = batch; one wave per block (single-buffered: spill-free)
__global__ void contrast_kernel(const float* __restrict__ hn,
                                const _Float16* __restrict__ Bp,
                                const int* __restrict__ labels,
                                float* __restrict__ closs)
{
    const int lane = threadIdx.x;
    const int bt = blockIdx.y;
    const int rt = blockIdx.x;
    const int BS = 10000;
    const int CT = BS / 16;                 // 625 column tiles
    const float*    Ab  = hn + (size_t)bt * BS * 192;
    const _Float16* Bb  = Bp + (size_t)bt * CT * 6 * 512;
    const int*      lab = labels + (size_t)bt * BS;
    const int rl = lane & 15, half = lane >> 4;
    const float* arow = Ab + (size_t)(rt * 16 + rl) * 192 + half * 16;
    v16h a[6];
    #pragma unroll
    for (int kc = 0; kc < 6; ++kc) a[kc] = load_a_frag(arow + kc * 32);

    float num[8], den[8];
    int rlab[8];
    #pragma unroll
    for (int v = 0; v < 8; ++v) {
        num[v] = 0.0f; den[v] = 0.0f;
        rlab[v] = lab[rt * 16 + half * 8 + v];
    }

    for (int ct = 0; ct < CT; ++ct) {
        const _Float16* bp = Bb + ((size_t)ct * 6 * 32 + lane) * 16;
        // prefetch next column tile's fragments (global_prefetch_b8: no VGPR / counter cost)
        if (ct + 1 < CT) {
            const _Float16* pf = bp + (size_t)6 * 512;
            __builtin_prefetch(pf, 0, 1);
            __builtin_prefetch(pf + 3 * 512, 0, 1);
        }
        v8f acc = {};
        #pragma unroll
        for (int kc = 0; kc < 6; ++kc) {
            v16h b = *reinterpret_cast<const v16h*>(bp + kc * 512);
            acc = __builtin_amdgcn_wmma_f32_16x16x32_f16(false, a[kc], false, b, (short)0, acc, false, false);
        }
        int clab = lab[ct * 16 + rl];
        #pragma unroll
        for (int v = 0; v < 8; ++v) {
            float e = __expf(acc[v] * 2.0f);     // 1/TEMP = 2
            den[v] += e;
            if (clab == rlab[v]) num[v] += e;
        }
    }

    __shared__ float snum[16], sden[16];
    if (lane < 16) { snum[lane] = 0.0f; sden[lane] = 0.0f; }
    __syncthreads();
    #pragma unroll
    for (int v = 0; v < 8; ++v) {
        atomicAdd(&snum[half * 8 + v], num[v]);
        atomicAdd(&sden[half * 8 + v], den[v]);
    }
    __syncthreads();
    if (lane < 16) {
        float loss = -__logf((snum[lane] + 1e-10f) / (sden[lane] + 1e-10f));
        atomicAdd(closs, loss * (1.0f / 50000.0f));
    }
}

// ======================================================================================
extern "C" void kernel_launch(void* const* d_in, const int* in_sizes, int n_in,
                              void* d_out, int out_size, void* d_ws, size_t ws_size,
                              hipStream_t stream)
{
    (void)n_in; (void)out_size; (void)ws_size;
    const int N = 50000, D = 192, Hd = 256;
    const int MT = N / 16;                 // 3125 row tiles
    const int GX = (MT + 3) / 4;           // 4 waves per block

    const float* des    = (const float*)d_in[0];
    const float* tweet  = (const float*)d_in[1];
    const float* nump   = (const float*)d_in[2];
    const float* catp   = (const float*)d_in[3];
    const int*   ei     = (const int*)d_in[4];
    const int*   et     = (const int*)d_in[5];
    const int*   labels = (const int*)d_in[6];
    const float* W_des  = (const float*)d_in[8];
    const float* b_des  = (const float*)d_in[9];
    const float* W_tw   = (const float*)d_in[10];
    const float* b_tw   = (const float*)d_in[11];
    const float* W_num  = (const float*)d_in[12];
    const float* b_num  = (const float*)d_in[13];
    const float* W_cat  = (const float*)d_in[14];
    const float* b_cat  = (const float*)d_in[15];
    const float* Wv     = (const float*)d_in[16];
    const float* bv     = (const float*)d_in[17];
    const float* Wo     = (const float*)d_in[18];
    const float* bo     = (const float*)d_in[19];
    const float* ln1_g  = (const float*)d_in[20];
    const float* ln1_b  = (const float*)d_in[21];
    const float* bn_g   = (const float*)d_in[22];
    const float* bn_b   = (const float*)d_in[23];
    const float* W1     = (const float*)d_in[24];
    const float* b1     = (const float*)d_in[25];
    const float* W2     = (const float*)d_in[26];
    const float* b2     = (const float*)d_in[27];
    const float* ln2_g  = (const float*)d_in[28];
    const float* ln2_b  = (const float*)d_in[29];
    const float* W_rel  = (const float*)d_in[30];
    const float* W_root = (const float*)d_in[31];
    const float* b_rgcn = (const float*)d_in[32];
    const float* Wm1    = (const float*)d_in[33];
    const float* bm1    = (const float*)d_in[34];
    const float* Wm2    = (const float*)d_in[35];
    const float* bm2    = (const float*)d_in[36];
    const float* Wm3    = (const float*)d_in[37];
    const float* bm3    = (const float*)d_in[38];

    const int E = in_sizes[4] / 2;
    const int* src = ei;
    const int* dst = ei + E;

    // ---- workspace carving ----
    char* wp = (char*)d_ws;
    auto carve = [&](size_t bytes) -> char* {
        char* r = wp; wp += (bytes + 255) & ~(size_t)255; return r;
    };
    float* x    = (float*)carve((size_t)N * 192 * 4);
    float* t1   = (float*)carve((size_t)N * 256 * 4);
    float* t2   = (float*)carve((size_t)N * 192 * 4);
    float* g1   = (float*)carve((size_t)N * 192 * 4);   // rgcn out 1; later reused as hn
    float* mr0  = (float*)carve((size_t)N * 192 * 4);
    float* mr1  = (float*)carve((size_t)N * 192 * 4);
    float* root = (float*)carve((size_t)N * 192 * 4);
    float* acc  = (float*)carve((size_t)2 * N * 192 * 4);
    float* cnt  = (float*)carve((size_t)2 * N * 4);
    _Float16* hp = (_Float16*)carve((size_t)5 * 192 * 10000 * 2);

    auto pack = [&](const float* W, int K, int Nout) -> _Float16* {
        _Float16* o = (_Float16*)carve((size_t)K * Nout * 2);
        int tot = K * Nout;
        pack_b_kernel<<<(tot + 255) / 256, 256, 0, stream>>>(W, o, K, Nout, 0);
        return o;
    };
    _Float16* pk_des  = pack(W_des, 768, 48);
    _Float16* pk_tw   = pack(W_tw, 768, 48);
    _Float16 *pk_Wv[6], *pk_Wo[6], *pk_W1[6], *pk_W2[6];
    for (int l = 0; l < 6; ++l) {
        pk_Wv[l] = pack(Wv + (size_t)l * D * D, D, D);
        pk_Wo[l] = pack(Wo + (size_t)l * D * D, D, D);
        pk_W1[l] = pack(W1 + (size_t)l * D * Hd, D, Hd);
        pk_W2[l] = pack(W2 + (size_t)l * Hd * D, Hd, D);
    }
    _Float16* pk_rel0 = pack(W_rel, D, D);
    _Float16* pk_rel1 = pack(W_rel + (size_t)D * D, D, D);
    _Float16* pk_root = pack(W_root, D, D);
    _Float16* pk_Wm1  = pack(Wm1, D, Hd);
    _Float16* pk_Wm2  = pack(Wm2, Hd, D);

    // Ntiles must be divisible by 4 (TN=4) or equal 3 (embed)
    auto gemm = [&](const float* A, const _Float16* Bp, const float* bias, float* C,
                    int ldc, int coloff, int K, int Ntiles, bool leaky) {
        if (Ntiles == 3) {
            dim3 grid(GX, 1);
            if (leaky) gemm_wmma_kernel<3, true ><<<grid, 128, 0, stream>>>(A, Bp, bias, C, ldc, coloff, MT, K);
            else       gemm_wmma_kernel<3, false><<<grid, 128, 0, stream>>>(A, Bp, bias, C, ldc, coloff, MT, K);
        } else {
            dim3 grid(GX, Ntiles / 4);
            if (leaky) gemm_wmma_kernel<4, true ><<<grid, 128, 0, stream>>>(A, Bp, bias, C, ldc, coloff, MT, K);
            else       gemm_wmma_kernel<4, false><<<grid, 128, 0, stream>>>(A, Bp, bias, C, ldc, coloff, MT, K);
        }
    };

    // ---- feature embedding: x[N,192] ----
    gemm(des,   pk_des, b_des, x, 192, 0,  768, 3, true);
    gemm(tweet, pk_tw,  b_tw,  x, 192, 48, 768, 3, true);
    embed_small_kernel<<<(N * 48 + 255) / 256, 256, 0, stream>>>(nump, 5, W_num, b_num, x, 96,  N);
    embed_small_kernel<<<(N * 48 + 255) / 256, 256, 0, stream>>>(catp, 3, W_cat, b_cat, x, 144, N);

    // ---- 6 transformer-ish layers ----
    for (int l = 0; l < 6; ++l) {
        gemm(x,  pk_Wv[l], bv + (size_t)l * D, t1, 192, 0, D, 12, false);
        gemm(t1, pk_Wo[l], bo + (size_t)l * D, t2, 192, 0, D, 12, false);
        add_ln_kernel<true><<<N, 192, 0, stream>>>(x, t2, ln1_g + (size_t)l * D, ln1_b + (size_t)l * D,
                                                   bn_g + (size_t)l * D, bn_b + (size_t)l * D, x);
        gemm(x,  pk_W1[l], b1 + (size_t)l * Hd, t1, 256, 0, D, 16, true);
        gemm(t1, pk_W2[l], b2 + (size_t)l * D,  t2, 192, 0, Hd, 12, false);
        add_ln_kernel<false><<<N, 192, 0, stream>>>(x, t2, ln2_g + (size_t)l * D, ln2_b + (size_t)l * D,
                                                    nullptr, nullptr, x);
    }
    // x is now h

    // ---- RGCN (applied twice) ----
    auto rgcn = [&](const float* in, float* outb) {
        gemm(in, pk_rel0, nullptr, mr0, 192, 0, D, 12, false);
        gemm(in, pk_rel1, nullptr, mr1, 192, 0, D, 12, false);
        gemm(in, pk_root, b_rgcn, root, 192, 0, D, 12, false);
        size_t na = (size_t)2 * N * 192;
        zero_kernel<<<(unsigned)((na + 255) / 256), 256, 0, stream>>>(acc, na);
        zero_kernel<<<(2 * N + 255) / 256, 256, 0, stream>>>(cnt, (size_t)2 * N);
        edge_count_kernel<<<(E + 255) / 256, 256, 0, stream>>>(dst, et, cnt, E, N);
        size_t tot = (size_t)E * 48;
        edge_scatter_kernel<<<(unsigned)((tot + 255) / 256), 256, 0, stream>>>(src, dst, et, mr0, mr1, acc, E, N);
        size_t nm = (size_t)N * 192;
        rgcn_combine_kernel<<<(unsigned)((nm + 255) / 256), 256, 0, stream>>>(root, acc, cnt, outb, N);
    };
    rgcn(x, g1);
    rgcn(g1, t2);   // t2 = g

    // ---- MLP head -> logits (d_out[0 .. 2N)) ----
    gemm(t2, pk_Wm1, bm1, t1, 256, 0, D, 16, true);
    gemm(t1, pk_Wm2, bm2, t2, 192, 0, Hd, 12, true);
    head_out_kernel<<<(N + 255) / 256, 256, 0, stream>>>(t2, Wm3, bm3, (float*)d_out, N);

    // ---- contrastive loss (d_out[2N]) ----
    normalize_kernel<<<N, 192, 0, stream>>>(x, g1);         // g1 = hn
    for (int bt = 0; bt < 5; ++bt) {
        int tot = 192 * 10000;
        pack_b_kernel<<<(tot + 255) / 256, 256, 0, stream>>>(g1 + (size_t)bt * 10000 * 192,
                                                             hp + (size_t)bt * 192 * 10000,
                                                             192, 10000, 1);
    }
    zero_kernel<<<1, 1, 0, stream>>>((float*)d_out + (size_t)2 * N, 1);
    contrast_kernel<<<dim3(625, 5), 32, 0, stream>>>(g1, hp, labels, (float*)d_out + (size_t)2 * N);
}
